// Model_GINE_15616501088841
// MI455X (gfx1250) — compile-verified, matrix operand order checked by reading, not run on hardware
//
#include <hip/hip_runtime.h>
#include <hip/hip_bf16.h>

typedef __attribute__((ext_vector_type(16))) _Float16 v16h;
typedef __attribute__((ext_vector_type(8)))  float    v8f;

#define NNODE   4096
#define TIME_D  64
#define NEDGE   131072
#define NBIN    129
#define DFFT    258
#define KPAD_ND 288   // node-MLP layer1 K pad (mult of 32)
#define NPAD_EE 272   // edge-MLP output N pad (mult of 16)

// ---------------------------------------------------------------------------
// WMMA helpers (CDNA5 gfx1250: V_WMMA_F32_16X16X32_F16, wave32)
// ---------------------------------------------------------------------------
__device__ __forceinline__ v8f wmma16(v16h a, v16h b, v8f c) {
  return __builtin_amdgcn_wmma_f32_16x16x32_f16(
      /*neg_a=*/false, a, /*neg_b=*/false, b,
      /*c_mod=*/(short)0, c, /*reuse_a=*/false, /*reuse_b=*/false);
}

union FragU { v16h v; unsigned u[8]; };

// Load a 16x32 fragment (A layout; B uses same layout on B^T) from a
// row-major tile with leading dimension ld (elements). Works for LDS or
// global pointers. Per ISA: lane<16 -> M=lane, K={0..7,16..23};
// lane>=16 -> M=lane-16, K={8..15,24..31}.  K pairs are contiguous and the
// 4 low pairs / 4 high pairs are 16B-contiguous -> compiler emits b128 loads.
__device__ __forceinline__ v16h load_frag(const _Float16* tile, int ld) {
  int lane = threadIdx.x & 31;
  int m    = lane & 15;
  int half = lane >> 4;
  const _Float16* p = tile + m * ld + 8 * half;
  FragU f;
#pragma unroll
  for (int i = 0; i < 8; ++i) {
    int kb = (i < 4) ? (2 * i) : (2 * i + 8);
    f.u[i] = *(const unsigned*)(p + kb);
  }
  return f.v;
}

// Branch-free tanh: keeps WMMA epilogues co-executable (no saveexec paths).
__device__ __forceinline__ float fast_tanh(float x) {
  x = fminf(fmaxf(x, -10.0f), 10.0f);
  float t = __expf(2.0f * x);
  return __fdividef(t - 1.0f, t + 1.0f);
}

template <int ACT> __device__ __forceinline__ float actf(float x) {
  if (ACT == 1) return fmaxf(x, 0.0f);
  if (ACT == 2) return fast_tanh(x);
  return x;
}

// ---------------------------------------------------------------------------
// Tensor Data Mover: stage a 16 x cols f16 tile (row stride = strideElems)
// from global into LDS.  2-D descriptor, data_size=2B, OOB rows zero-fill.
// Arity differs across toolchains (probe doc): clang-22 = 5 args, clang-23 = 6.
// ---------------------------------------------------------------------------
#if __has_builtin(__builtin_amdgcn_tensor_load_to_lds)
#define HAVE_TDM 1
typedef __attribute__((ext_vector_type(4))) unsigned tdm_v4u;
typedef __attribute__((ext_vector_type(8))) int      tdm_v8i;
typedef __attribute__((ext_vector_type(4))) int      tdm_v4i;

__device__ __forceinline__ void tdm_load_tile16(const _Float16* gsrc,
                                                unsigned ldsByteAddr,
                                                int rowsValid, int cols,
                                                int strideElems) {
  unsigned long long ga = (unsigned long long)(size_t)gsrc;
  tdm_v4u g0;
  g0[0] = 1u;                                     // count=1 (valid user D#)
  g0[1] = ldsByteAddr;                            // lds_addr
  g0[2] = (unsigned)(ga & 0xffffffffu);           // global_addr[31:0]
  g0[3] = (unsigned)((ga >> 32) & 0x01ffffffu)    // global_addr[56:32]
        | (2u << 30);                             // type=2 ("image")
  tdm_v8i g1;
  unsigned d0 = (unsigned)cols, d1 = (unsigned)rowsValid, st = (unsigned)strideElems;
  g1[0] = (int)(1u << 16);                        // wg_mask=0, data_size=1 (2B)
  g1[1] = (int)(d0 << 16);                        // tensor_dim0[15:0] @ [63:48]
  g1[2] = (int)((d0 >> 16) | (d1 << 16));         // dim0[31:16] | dim1[15:0]
  g1[3] = (int)((d1 >> 16) | (d0 << 16));         // dim1[31:16] | tile_dim0
  g1[4] = 16;                                     // tile_dim1=16, tile_dim2=0
  g1[5] = (int)st;                                // tensor_dim0_stride[31:0]
  g1[6] = 0;                                      // stride[47:32], dim1_stride
  g1[7] = 0;
  tdm_v4i z = {0, 0, 0, 0};
#if __clang_major__ >= 23
  tdm_v8i z8 = {0, 0, 0, 0, 0, 0, 0, 0};
  __builtin_amdgcn_tensor_load_to_lds(g0, g1, z, z, z8, 0);
#else
  __builtin_amdgcn_tensor_load_to_lds(g0, g1, z, z, 0);
#endif
}
#endif

// ---------------------------------------------------------------------------
// Weight prep: f32 W[K][N] -> f16 Bt[Npad][Kpad] (zero padded)
// ---------------------------------------------------------------------------
__global__ void convert_wt(const float* __restrict__ W, _Float16* __restrict__ out,
                           int K, int N, int Kpad, int Npad) {
  int i = blockIdx.x * 256 + threadIdx.x;
  if (i >= Kpad * Npad) return;
  int n = i / Kpad, k = i % Kpad;
  float v = (n < N && k < K) ? W[(size_t)k * N + n] : 0.0f;
  out[i] = (_Float16)v;
}

__global__ void f32_to_f16(const float* __restrict__ in, _Float16* __restrict__ out, int n) {
  int i = blockIdx.x * 256 + threadIdx.x;
  if (i < n) out[i] = (_Float16)in[i];
}

// ---------------------------------------------------------------------------
// rFFT(64 samples zero-padded to 256) -> [abs(129) | angle(129)] per node.
// Also initializes hg = node_fft (GINE: hg = node_fft + scatter_sum later).
// ---------------------------------------------------------------------------
__global__ void fft_kernel(const float* __restrict__ accY,
                           float* __restrict__ node_fft, float* __restrict__ hg) {
  int i = blockIdx.x * 256 + threadIdx.x;
  if (i >= NNODE * NBIN) return;
  int node = i / NBIN, k = i % NBIN;
  float w = (6.283185307179586f / 256.0f) * (float)k;
  const float* x = accY + (size_t)node * TIME_D;
  float re = 0.0f, im = 0.0f;
  for (int t = 0; t < TIME_D; ++t) {
    float s, c;
    __sincosf(w * (float)t, &s, &c);
    float v = x[t];
    re += v * c;
    im -= v * s;
  }
  float ab = sqrtf(re * re + im * im);
  float an = atan2f(im, re);
  size_t base = (size_t)node * DFFT;
  node_fft[base + k] = ab;        hg[base + k] = ab;
  node_fft[base + NBIN + k] = an; hg[base + NBIN + k] = an;
}

// ---------------------------------------------------------------------------
// Generic WMMA GEMM: out[M][Npad](f16) = ACT(A[M][Kpad](f16) @ Bt^T + bias)
// Bt is f16 [Npad][Kpad].  Block = 128 threads (4 waves), 16 rows per block.
// A tile staged to LDS by the Tensor Data Mover (fallback: manual copy).
// ---------------------------------------------------------------------------
template <int ACT>
__global__ __launch_bounds__(128) void gemm16(
    const _Float16* __restrict__ A, const _Float16* __restrict__ Bt,
    const float* __restrict__ bias, _Float16* __restrict__ out,
    int M, int Kpad, int Npad) {
  __shared__ _Float16 aLds[16 * 256];
  int tid = threadIdx.x;
  int r0  = blockIdx.x * 16;
#if defined(HAVE_TDM)
  if (tid < 32) {  // wave 0 drives the TDM; OOB rows (>=M) zero-fill
    tdm_load_tile16(A + (size_t)r0 * Kpad, (unsigned)(size_t)aLds,
                    M - r0, Kpad, Kpad);
    __builtin_amdgcn_s_wait_tensorcnt(0);
  }
#else
  {
    int kp2 = Kpad >> 1;
    const unsigned* Ag = (const unsigned*)A;
    unsigned* aw = (unsigned*)aLds;
    for (int i = tid; i < 16 * kp2; i += 128) {
      int row = i / kp2, col = i - row * kp2;
      unsigned v = 0;
      if (r0 + row < M) v = Ag[(size_t)(r0 + row) * kp2 + col];
      aw[row * kp2 + col] = v;
    }
  }
#endif
  __syncthreads();
  int w = tid >> 5, lane = tid & 31, n = lane & 15, half = lane >> 4;
  int ntiles = Npad >> 4;
  for (int nt = w; nt < ntiles; nt += 4) {
    v8f acc = {};
    for (int k0 = 0; k0 < Kpad; k0 += 32) {
      v16h a = load_frag(aLds + k0, Kpad);
      v16h b = load_frag(Bt + (size_t)nt * 16 * Kpad + k0, Kpad);
      acc = wmma16(a, b, acc);
    }
    int c = nt * 16 + n;
    float bb = bias ? bias[c] : 0.0f;
#pragma unroll
    for (int r = 0; r < 8; ++r) {
      int row = half * 8 + r;
      if (r0 + row < M)
        out[(size_t)(r0 + row) * Npad + c] = (_Float16)actf<ACT>(acc[r] + bb);
    }
  }
}

// ---------------------------------------------------------------------------
// Attention (H=4, DH=4): online-softmax, one thread per (query, head).
// q16 [Mq][16], k16/v16 [Mk][16], o16 out [Mq][32] (cols 16..31 zeroed so it
// can feed a Kpad=32 WMMA GEMM for Wo).
// ---------------------------------------------------------------------------
__global__ __launch_bounds__(128) void attn_kernel(
    const _Float16* __restrict__ q16, const _Float16* __restrict__ k16,
    const _Float16* __restrict__ v16, _Float16* __restrict__ o16, int Mq, int Mk) {
  __shared__ _Float16 kl[64 * 16];
  __shared__ _Float16 vl[64 * 16];
  int tid = threadIdx.x;
  int qi = blockIdx.x * 32 + (tid >> 2);
  int h  = tid & 3;
  float qv[4] = {0, 0, 0, 0};
  if (qi < Mq)
#pragma unroll
    for (int d = 0; d < 4; ++d) qv[d] = (float)q16[(size_t)qi * 16 + h * 4 + d];
  float m = -1e30f, l = 0.0f, acc[4] = {0, 0, 0, 0};
  for (int j0 = 0; j0 < Mk; j0 += 64) {
    int cnt = (Mk - j0 < 64) ? (Mk - j0) : 64;
    for (int i = tid; i < 64 * 8; i += 128) {  // uint copies (2 halfs each)
      int row = i >> 3, cu = i & 7;
      unsigned a = 0, b = 0;
      if (row < cnt) {
        a = ((const unsigned*)(k16 + (size_t)(j0 + row) * 16))[cu];
        b = ((const unsigned*)(v16 + (size_t)(j0 + row) * 16))[cu];
      }
      ((unsigned*)kl)[i] = a;
      ((unsigned*)vl)[i] = b;
    }
    __syncthreads();
    if (qi < Mq) {
      for (int j = 0; j < cnt; ++j) {
        float s = 0.0f;
#pragma unroll
        for (int d = 0; d < 4; ++d) s += qv[d] * (float)kl[j * 16 + h * 4 + d];
        s *= 0.5f;  // DH^-0.5
        float mn = fmaxf(m, s);
        float cr = __expf(m - mn);
        float p  = __expf(s - mn);
        l = l * cr + p;
#pragma unroll
        for (int d = 0; d < 4; ++d)
          acc[d] = acc[d] * cr + p * (float)vl[j * 16 + h * 4 + d];
        m = mn;
      }
    }
    __syncthreads();
  }
  if (qi < Mq) {
    float inv = 1.0f / l;
#pragma unroll
    for (int d = 0; d < 4; ++d) {
      o16[(size_t)qi * 32 + h * 4 + d]      = (_Float16)(acc[d] * inv);
      o16[(size_t)qi * 32 + 16 + h * 4 + d] = (_Float16)0.0f;
    }
  }
}

// LayerNorm(a + b) over D=64, f32 gamma/beta, f16 in/out. One row per thread.
__global__ void ln_kernel(const _Float16* __restrict__ a, const _Float16* __restrict__ b,
                          const float* __restrict__ g, const float* __restrict__ be,
                          _Float16* __restrict__ out, int M) {
  int m = blockIdx.x * 256 + threadIdx.x;
  if (m >= M) return;
  float s = 0.0f, s2 = 0.0f;
  for (int d = 0; d < 64; ++d) {
    float x = (float)a[(size_t)m * 64 + d] + (float)b[(size_t)m * 64 + d];
    s += x; s2 += x * x;
  }
  float mean = s * (1.0f / 64.0f);
  float var  = s2 * (1.0f / 64.0f) - mean * mean;
  float inv  = rsqrtf(var + 1e-5f);
  for (int d = 0; d < 64; ++d) {
    float x = (float)a[(size_t)m * 64 + d] + (float)b[(size_t)m * 64 + d];
    out[(size_t)m * 64 + d] = (_Float16)((x - mean) * inv * g[d] + be[d]);
  }
}

// Shared WMMA layer: outLds[.][128] = tanh(in @ Bt + bias), K=Kpad.
__device__ __forceinline__ void wmma_layer_tanh(
    const _Float16* in, _Float16* outLds, const _Float16* __restrict__ Bt,
    const float* __restrict__ bias, int mt, int ntBegin, int ntStride, int Kpad) {
  int lane = threadIdx.x & 31, n = lane & 15, half = lane >> 4;
  for (int nt = ntBegin; nt < 8; nt += ntStride) {
    v8f acc = {};
    for (int k0 = 0; k0 < Kpad; k0 += 32) {
      v16h a = load_frag(in + mt * 16 * Kpad + k0, Kpad);
      v16h b = load_frag(Bt + (size_t)nt * 16 * Kpad + k0, Kpad);
      acc = wmma16(a, b, acc);
    }
    float bb = bias[nt * 16 + n];
#pragma unroll
    for (int r = 0; r < 8; ++r)
      outLds[(mt * 16 + half * 8 + r) * 128 + nt * 16 + n] =
          (_Float16)fast_tanh(acc[r] + bb);
  }
}

// ---------------------------------------------------------------------------
// Fused edge MLP (2->128->128->128->258, tanh) + GINE scatter:
//   hg[dst] += relu(node_fft[src] + edge_enc).  64 edges per block (4 waves),
//   activations live in LDS; weights are f16 in L2 (192MB -> resident).
// ---------------------------------------------------------------------------
__global__ __launch_bounds__(128) void edge_mlp_kernel(
    const float* __restrict__ Lw, const float* __restrict__ theta,
    const int* __restrict__ src, const int* __restrict__ dst,
    const float* __restrict__ W1, const float* __restrict__ b1,
    const _Float16* __restrict__ W2t, const float* __restrict__ b2,
    const _Float16* __restrict__ W3t, const float* __restrict__ b3,
    const float* __restrict__ node_fft, float* __restrict__ hg) {
  __shared__ _Float16 h0[64 * 128];
  __shared__ _Float16 h1[64 * 128];
  __shared__ float sx0[64], sx1[64];
  __shared__ int ssrc[64], sdst[64];
  int tid = threadIdx.x;
  int e0  = blockIdx.x * 64;
  if (tid < 64) {
    int e = e0 + tid;
    sx0[tid] = Lw[e];
    sx1[tid] = theta[e] * (1.0f / 360.0f);
    ssrc[tid] = src[e];
    sdst[tid] = dst[e];
  }
  __syncthreads();
  {  // layer1: K=2, pure VALU
    int c = tid;
    float w0 = W1[c], w1 = W1[128 + c], bb = b1[c];
    for (int e = 0; e < 64; ++e)
      h0[e * 128 + c] = (_Float16)fast_tanh(sx0[e] * w0 + sx1[e] * w1 + bb);
  }
  __syncthreads();
  int w = tid >> 5, lane = tid & 31, n = lane & 15, half = lane >> 4;
  wmma_layer_tanh(h0, h1, W2t, b2, w, 0, 1, 128);  // layer2
  __syncthreads();
  wmma_layer_tanh(h1, h0, W2t, b2, w, 0, 1, 128);  // layer3
  __syncthreads();
  // layer4 (128 -> 272 padded, 258 valid) fused with GINE gather/relu/scatter
  for (int nt = 0; nt < 17; ++nt) {
    if (nt + 1 < 17) __builtin_prefetch(W3t + (size_t)(nt + 1) * 16 * 128, 0, 0);
    v8f acc = {};
    for (int k0 = 0; k0 < 128; k0 += 32) {
      v16h a = load_frag(h0 + w * 16 * 128 + k0, 128);
      v16h b = load_frag(W3t + (size_t)nt * 16 * 128 + k0, 128);
      acc = wmma16(a, b, acc);
    }
    int c = nt * 16 + n;
    if (c < DFFT) {
      float bb = b3[c];
#pragma unroll
      for (int r = 0; r < 8; ++r) {
        int el = w * 16 + half * 8 + r;
        int s = ssrc[el], d = sdst[el];
        float v = node_fft[(size_t)s * DFFT + c] + acc[r] + bb;
        v = fmaxf(v, 0.0f);
        atomicAdd(hg + (size_t)d * DFFT + c, v);
      }
    }
  }
}

// ---------------------------------------------------------------------------
// Fused node MLP (258->128->128->128->8, tanh, linear out). 32 nodes/block.
// ---------------------------------------------------------------------------
__global__ __launch_bounds__(128) void node_mlp_kernel(
    const float* __restrict__ hg,
    const _Float16* __restrict__ W1t, const float* __restrict__ b1,
    const _Float16* __restrict__ W2t, const float* __restrict__ b2,
    const _Float16* __restrict__ W3t, const float* __restrict__ b3,
    float* __restrict__ phi_raw) {
  __shared__ _Float16 abuf[32 * KPAD_ND];
  __shared__ _Float16 h0[32 * 128];
  __shared__ _Float16 h1[32 * 128];
  int tid = threadIdx.x;
  int n0  = blockIdx.x * 32;
  for (int i = tid; i < 32 * KPAD_ND; i += 128) {
    int row = i / KPAD_ND, col = i - row * KPAD_ND;
    float v = (col < DFFT) ? hg[(size_t)(n0 + row) * DFFT + col] : 0.0f;
    abuf[i] = (_Float16)v;
  }
  __syncthreads();
  int w = tid >> 5, lane = tid & 31, n = lane & 15, half = lane >> 4;
  int mt = w & 1, ws = w >> 1;  // 2 waves per 16-row tile, split over N-tiles
  wmma_layer_tanh(abuf, h0, W1t, b1, mt, ws, 2, KPAD_ND);  // layer1 K=288
  __syncthreads();
  wmma_layer_tanh(h0, h1, W2t, b2, mt, ws, 2, 128);        // layer2
  __syncthreads();
  wmma_layer_tanh(h1, h0, W2t, b2, mt, ws, 2, 128);        // layer3
  __syncthreads();
  if (ws == 0) {  // layer4: 128 -> 16 (8 valid), linear, f32 out
    v8f acc = {};
    for (int k0 = 0; k0 < 128; k0 += 32) {
      v16h a = load_frag(h0 + mt * 16 * 128 + k0, 128);
      v16h b = load_frag(W3t + k0, 128);
      acc = wmma16(a, b, acc);
    }
    if (n < 8) {
      float bb = b3[n];
#pragma unroll
      for (int r = 0; r < 8; ++r)
        phi_raw[(size_t)(n0 + mt * 16 + half * 8 + r) * 8 + n] = acc[r] + bb;
    }
  }
}

__global__ void colmax_kernel(const float* __restrict__ phi, float* __restrict__ maxv) {
  __shared__ float red[256];
  int c = blockIdx.x;
  float mx = 0.0f;
  for (int r = threadIdx.x; r < NNODE; r += 256)
    mx = fmaxf(mx, fabsf(phi[(size_t)r * 8 + c]));
  red[threadIdx.x] = mx;
  __syncthreads();
  for (int s = 128; s > 0; s >>= 1) {
    if (threadIdx.x < s) red[threadIdx.x] = fmaxf(red[threadIdx.x], red[threadIdx.x + s]);
    __syncthreads();
  }
  if (threadIdx.x == 0) maxv[c] = red[0];
}

__global__ void phi_out_kernel(const float* __restrict__ phi, const float* __restrict__ maxv,
                               float* __restrict__ out) {
  int i = blockIdx.x * 256 + threadIdx.x;
  if (i < NNODE * 8) out[512 + i] = phi[i] / maxv[i & 7];
}

__global__ void s_out_kernel(const _Float16* __restrict__ s, float* __restrict__ out) {
  int i = blockIdx.x * 256 + threadIdx.x;
  if (i < 512) out[i] = (float)s[i];
}

// ---------------------------------------------------------------------------
// Host orchestration
// ---------------------------------------------------------------------------
extern "C" void kernel_launch(void* const* d_in, const int* in_sizes, int n_in,
                              void* d_out, int out_size, void* d_ws, size_t ws_size,
                              hipStream_t stream) {
  (void)in_sizes; (void)n_in; (void)out_size; (void)ws_size;
  char* ws = (char*)d_ws;
  float* out = (float*)d_out;
  auto fp  = [&](int i) { return (const float*)d_in[i]; };
  auto ip  = [&](int i) { return (const int*)d_in[i]; };
  size_t o = 0;
  auto alloc = [&](size_t bytes) { size_t r = o; o += (bytes + 255) & ~(size_t)255; return r; };
  auto F16 = [&](size_t off) { return (_Float16*)(ws + off); };
  auto F32 = [&](size_t off) { return (float*)(ws + off); };

  // f16 weight buffers ([Npad][Kpad] layout)
  size_t wq[5], wk[5], wv[5], wo[5], wf1[5], wf2[5];
  for (int b = 0; b < 5; ++b) {
    wq[b]  = alloc(16 * 64 * 2);  wk[b]  = alloc(16 * 64 * 2);  wv[b] = alloc(16 * 64 * 2);
    wo[b]  = alloc(64 * 32 * 2);  wf1[b] = alloc(256 * 64 * 2); wf2[b] = alloc(64 * 256 * 2);
  }
  size_t pmaf1 = alloc(256 * 64 * 2), pmaf2 = alloc(64 * 256 * 2);
  size_t eeW2t = alloc(128 * 128 * 2), eeW3t = alloc((size_t)NPAD_EE * 128 * 2);
  size_t ndW1t = alloc(128 * KPAD_ND * 2), ndW2t = alloc(128 * 128 * 2), ndW3t = alloc(16 * 128 * 2);
  // activations (f16)
  size_t xb = alloc((size_t)NNODE * 64 * 2), yb = alloc((size_t)NNODE * 64 * 2);
  size_t qb = alloc((size_t)NNODE * 16 * 2), kb = alloc((size_t)NNODE * 16 * 2);
  size_t vb = alloc((size_t)NNODE * 16 * 2), ab = alloc((size_t)NNODE * 32 * 2);
  size_t t64a = alloc((size_t)NNODE * 64 * 2), t256 = alloc((size_t)NNODE * 256 * 2);
  size_t t64b = alloc((size_t)NNODE * 64 * 2);
  size_t sxo = alloc(8 * 64 * 2);
  // f32 buffers
  size_t nfft = alloc((size_t)NNODE * DFFT * 4), hgo = alloc((size_t)NNODE * DFFT * 4);
  size_t phio = alloc((size_t)NNODE * 8 * 4), mxo = alloc(8 * 4);

  // --- weight conversion -------------------------------------------------
  auto cw = [&](int pidx, size_t dst, int K, int N, int Kp, int Np) {
    int tot = Kp * Np;
    convert_wt<<<(tot + 255) / 256, 256, 0, stream>>>(fp(pidx), F16(dst), K, N, Kp, Np);
  };
  for (int b = 0; b < 5; ++b) {
    int pb = 5 + b * 12;
    cw(pb + 0, wq[b], 64, 16, 64, 16);
    cw(pb + 1, wk[b], 64, 16, 64, 16);
    cw(pb + 2, wv[b], 64, 16, 64, 16);
    cw(pb + 3, wo[b], 16, 64, 32, 64);
    cw(pb + 4, wf1[b], 64, 256, 64, 256);
    cw(pb + 6, wf2[b], 256, 64, 256, 64);
  }
  cw(66, pmaf1, 64, 256, 64, 256);
  cw(68, pmaf2, 256, 64, 256, 64);
  cw(72, eeW2t, 128, 128, 128, 128);
  cw(74, eeW3t, 128, DFFT, 128, NPAD_EE);
  cw(76, ndW1t, DFFT, 128, KPAD_ND, 128);
  cw(78, ndW2t, 128, 128, 128, 128);
  cw(80, ndW3t, 128, 8, 128, 16);

  f32_to_f16<<<(NNODE * 64 + 255) / 256, 256, 0, stream>>>(fp(0), F16(xb), NNODE * 64);
  f32_to_f16<<<2, 256, 0, stream>>>(fp(65), F16(sxo), 512);  // pma_seed

  // --- FFT features + hg init, then fused edge MLP + GINE scatter --------
  fft_kernel<<<(NNODE * NBIN + 255) / 256, 256, 0, stream>>>(fp(0), F32(nfft), F32(hgo));
  edge_mlp_kernel<<<NEDGE / 64, 128, 0, stream>>>(
      fp(1), fp(2), ip(3), ip(4), fp(70), fp(71), F16(eeW2t), fp(73),
      F16(eeW3t), fp(75), F32(nfft), F32(hgo));

  // --- SAB / MHA block (generic over x/mem sizes) ------------------------
  auto sab = [&](int b, _Float16* x, const _Float16* mem, int Mx, int Mm) {
    int pb = 5 + b * 12;
    int gx = (Mx + 15) / 16, gm = (Mm + 15) / 16;
    gemm16<0><<<gx, 128, 0, stream>>>(x,   F16(wq[b]), nullptr, F16(qb), Mx, 64, 16);
    gemm16<0><<<gm, 128, 0, stream>>>(mem, F16(wk[b]), nullptr, F16(kb), Mm, 64, 16);
    gemm16<0><<<gm, 128, 0, stream>>>(mem, F16(wv[b]), nullptr, F16(vb), Mm, 64, 16);
    attn_kernel<<<(Mx + 31) / 32, 128, 0, stream>>>(F16(qb), F16(kb), F16(vb), F16(ab), Mx, Mm);
    gemm16<0><<<gx, 128, 0, stream>>>(F16(ab), F16(wo[b]), nullptr, F16(t64a), Mx, 32, 64);
    ln_kernel<<<(Mx + 255) / 256, 256, 0, stream>>>(x, F16(t64a), fp(pb + 8), fp(pb + 9), F16(yb), Mx);
    gemm16<1><<<gx, 128, 0, stream>>>(F16(yb), F16(wf1[b]), fp(pb + 5), F16(t256), Mx, 64, 256);
    gemm16<0><<<gx, 128, 0, stream>>>(F16(t256), F16(wf2[b]), fp(pb + 7), F16(t64b), Mx, 256, 64);
    ln_kernel<<<(Mx + 255) / 256, 256, 0, stream>>>(F16(yb), F16(t64b), fp(pb + 10), fp(pb + 11), x, Mx);
  };

  // encoder: 2 SAB over all nodes
  sab(0, F16(xb), F16(xb), NNODE, NNODE);
  sab(1, F16(xb), F16(xb), NNODE, NNODE);

  // PMA: feat = relu(h@fW1+b1)@fW2+b2, then seeds attend over feat
  gemm16<1><<<NNODE / 16, 128, 0, stream>>>(F16(xb), F16(pmaf1), fp(67), F16(t256), NNODE, 64, 256);
  gemm16<0><<<NNODE / 16, 128, 0, stream>>>(F16(t256), F16(pmaf2), fp(69), F16(t64a), NNODE, 256, 64);
  sab(2, F16(sxo), F16(t64a), 8, NNODE);
  // decoder: 2 SAB over 8 seeds
  sab(3, F16(sxo), F16(sxo), 8, 8);
  sab(4, F16(sxo), F16(sxo), 8, 8);
  s_out_kernel<<<2, 256, 0, stream>>>(F16(sxo), out);  // q (1,64,8) flat

  // node MLP -> phi, then per-mode normalization
  node_mlp_kernel<<<NNODE / 32, 128, 0, stream>>>(
      F32(hgo), F16(ndW1t), fp(77), F16(ndW2t), fp(79), F16(ndW3t), fp(81), F32(phio));
  colmax_kernel<<<8, 256, 0, stream>>>(F32(phio), F32(mxo));
  phi_out_kernel<<<(NNODE * 8 + 255) / 256, 256, 0, stream>>>(F32(phio), F32(mxo), out);
}
